// NMTindoorLocPT_36034775613501
// MI455X (gfx1250) — matrix operand order repaired and checked by reading, not compiled, not run on hardware
//
#include <hip/hip_runtime.h>

typedef __bf16 bf16_t;
typedef __attribute__((ext_vector_type(16))) __bf16 v16bf;
typedef __attribute__((ext_vector_type(8)))  float  v8f;

#define BB   1024
#define TT   512
#define NI   6
#define NA   188
#define HH   256
#define OT   10
#define NCC  2
#define G4   1024          // 4*H
#define XIN  194           // NI + NA
#define XK   224           // x-path K padded to 32-multiple
#define XS   232           // LDS stride (elems) for x staging (bank-friendly, 8-mult)
#define HS   264           // LDS stride for h staging
#define DK   288           // decoder K = 256 + 2, padded
#define AS   296           // LDS stride for decoder A staging
#define BTILE 32
#define NWG  (BB / BTILE)  // 32 workgroups, batch-independent recurrences

union Frag { v16bf v; uint4 q[2]; };

__device__ __forceinline__ float sigf(float x) { return 1.0f / (1.0f + __expf(-x)); }

// ---------------------------------------------------------------------------
// Kernel 1: fold/convert weights into bf16 WMMA panels (zero-padded K).
//   WxP  [1024][224]  cols 0..5 = W_ih[:, :6], cols 6..193 = W_ih[:,6:] @ W_rss
//   Whh  [1024][256]  bf16(W_hh)
//   Wdec [1024][288]  cols 0..255 = Wc_hh, 256..257 = Wc_ih
//   beff [1024] = b_ih + b_hh + W_ih[:,6:] @ b_rss ;  bdec = bc_ih + bc_hh
// ---------------------------------------------------------------------------
__global__ void prep_kernel(const float* __restrict__ W_rss, const float* __restrict__ b_rss,
                            const float* __restrict__ W_ih,  const float* __restrict__ W_hh,
                            const float* __restrict__ b_ih,  const float* __restrict__ b_hh,
                            const float* __restrict__ Wc_ih, const float* __restrict__ Wc_hh,
                            const float* __restrict__ bc_ih, const float* __restrict__ bc_hh,
                            bf16_t* __restrict__ WxP, bf16_t* __restrict__ Whh,
                            bf16_t* __restrict__ Wdec, float* __restrict__ beff,
                            float* __restrict__ bdec) {
    const int n   = blockIdx.x;           // 0..1023 (gate row)
    const int tid = threadIdx.x;          // 256 threads
    const float* wi = W_ih + (size_t)n * (NI + HH);

    if (tid < XK) {
        float v = 0.0f;
        if (tid < NI) {
            v = wi[tid];
        } else if (tid < XIN) {
            const int a = tid - NI;
            for (int m = 0; m < HH; ++m) v += wi[NI + m] * W_rss[m * NA + a];
        }
        WxP[(size_t)n * XK + tid] = (bf16_t)v;
    }
    if (tid < HH) Whh[(size_t)n * HH + tid] = (bf16_t)W_hh[(size_t)n * HH + tid];
    for (int k = tid; k < DK; k += blockDim.x) {
        float v = 0.0f;
        if (k < HH)            v = Wc_hh[(size_t)n * HH + k];
        else if (k < HH + NCC) v = Wc_ih[(size_t)n * NCC + (k - HH)];
        Wdec[(size_t)n * DK + k] = (bf16_t)v;
    }
    if (tid == 0) {
        float s = b_ih[n] + b_hh[n];
        for (int m = 0; m < HH; ++m) s += wi[NI + m] * b_rss[m];
        beff[n] = s;
        bdec[n] = bc_ih[n] + bc_hh[n];
    }
}

// ---------------------------------------------------------------------------
// Kernel 2: encoder LSTM. One WG = 32 batch rows, all 512 timesteps.
// 16 waves: wave w -> j-slice j0 = w*16 (disjoint -> no dup weight fetch),
// covering BOTH 16-row batch sub-tiles and all 4 gates:
//   acc[mt*4+g] at N = g*256 + j0, M-rows = mt*16..mt*16+15.
// Each B-frag (weights) feeds 2 WMMAs; i/f/g/o are wave-local so c stays in
// registers; only bf16 h crosses waves via LDS (2 barriers/step).
// Weight loads are laundered through an opaque zero offset each step so LICM
// cannot hoist+spill them: they stream from the L2-resident bf16 panels.
// ---------------------------------------------------------------------------
__global__ void __launch_bounds__(512)
encoder_kernel(const float* __restrict__ imu, const float* __restrict__ rss,
               const bf16_t* __restrict__ WxP, const bf16_t* __restrict__ Whh,
               const float* __restrict__ beff,
               bf16_t* __restrict__ hfin, float* __restrict__ cfin) {
    __shared__ __align__(16) bf16_t xs[BTILE * XS];   // x_t staging (bf16, K-padded)
    __shared__ __align__(16) bf16_t hs[BTILE * HS];   // h_t staging (bf16)

    const int tid  = threadIdx.x;
    const int b0   = blockIdx.x * BTILE;
    const int lane = tid & 31;
    const int wv   = tid >> 5;
    const int j0   = wv * 16;            // 16 waves x 16 cols = 256 hidden cols
    const int c16  = lane & 15;
    const int hi8  = (lane >> 4) * 8;    // A- and B-frag k offset (same for both)

    // zero LDS (h0 = 0, x pad cols = 0)
    for (int i = tid; i < BTILE * XS; i += blockDim.x) xs[i] = (bf16_t)0.0f;
    for (int i = tid; i < BTILE * HS; i += blockDim.x) hs[i] = (bf16_t)0.0f;
    __syncthreads();

    // stage x_0
    for (int idx = tid; idx < BTILE * XIN; idx += blockDim.x) {
        const int r = idx / XIN, cc = idx - r * XIN;
        const int b = b0 + r;
        const float v = (cc < NI) ? imu[((size_t)b * TT) * NI + cc]
                                  : rss[((size_t)b * TT) * NA + (cc - NI)];
        xs[r * XS + cc] = (bf16_t)v;
    }
    __syncthreads();

    // per-gate bias (depends only on N column) — same for both batch sub-tiles
    float biasv[4];
#pragma unroll
    for (int g = 0; g < 4; ++g) biasv[g] = beff[g * HH + j0 + c16];

    v8f creg[2];
#pragma unroll
    for (int mt = 0; mt < 2; ++mt)
#pragma unroll
        for (int e = 0; e < 8; ++e) creg[mt][e] = 0.0f;

    unsigned zoff = 0;                    // opaque zero: defeats LICM of B loads

    for (int t = 0; t < TT; ++t) {
        asm volatile("" : "+s"(zoff));    // weights "may have moved" this step

        v8f acc[8];
#pragma unroll
        for (int mt = 0; mt < 2; ++mt)
#pragma unroll
            for (int g = 0; g < 4; ++g)
#pragma unroll
                for (int e = 0; e < 8; ++e) acc[mt * 4 + g][e] = biasv[g];

        // phase 1: x-path, K = 224 (7 WMMA k-steps x 8 WMMAs)
        for (int k0 = 0; k0 < XK; k0 += 32) {
            Frag fa[2];
#pragma unroll
            for (int mt = 0; mt < 2; ++mt) {
                const bf16_t* pa = &xs[(mt * 16 + c16) * XS + k0 + hi8];
                fa[mt].q[0] = *(const uint4*)pa;
                fa[mt].q[1] = *(const uint4*)(pa + 16);
            }
            Frag fb[4];
#pragma unroll
            for (int g = 0; g < 4; ++g) {
                const bf16_t* pb = WxP + zoff +
                    (size_t)(g * HH + j0 + c16) * XK + k0 + hi8;
                fb[g].q[0] = *(const uint4*)pb;
                fb[g].q[1] = *(const uint4*)(pb + 16);
            }
#pragma unroll
            for (int g = 0; g < 4; ++g) {
#pragma unroll
                for (int mt = 0; mt < 2; ++mt)
                    acc[mt * 4 + g] = __builtin_amdgcn_wmma_f32_16x16x32_bf16(
                        false, fa[mt].v, false, fb[g].v, (short)0,
                        acc[mt * 4 + g], false, false);
            }
        }
        // phase 2: recurrent path, K = 256 (8 WMMA k-steps x 8 WMMAs)
        for (int k0 = 0; k0 < HH; k0 += 32) {
            Frag fa[2];
#pragma unroll
            for (int mt = 0; mt < 2; ++mt) {
                const bf16_t* pa = &hs[(mt * 16 + c16) * HS + k0 + hi8];
                fa[mt].q[0] = *(const uint4*)pa;
                fa[mt].q[1] = *(const uint4*)(pa + 16);
            }
            Frag fb[4];
#pragma unroll
            for (int g = 0; g < 4; ++g) {
                const bf16_t* pb = Whh + zoff +
                    (size_t)(g * HH + j0 + c16) * HH + k0 + hi8;
                fb[g].q[0] = *(const uint4*)pb;
                fb[g].q[1] = *(const uint4*)(pb + 16);
            }
#pragma unroll
            for (int g = 0; g < 4; ++g) {
#pragma unroll
                for (int mt = 0; mt < 2; ++mt)
                    acc[mt * 4 + g] = __builtin_amdgcn_wmma_f32_16x16x32_bf16(
                        false, fa[mt].v, false, fb[g].v, (short)0,
                        acc[mt * 4 + g], false, false);
            }
        }
        __syncthreads();   // all waves done reading xs/hs for step t

        // elementwise LSTM update: c in registers, h -> LDS (bf16)
#pragma unroll
        for (int mt = 0; mt < 2; ++mt) {
            const v8f gi = acc[mt * 4 + 0], gf = acc[mt * 4 + 1];
            const v8f gg = acc[mt * 4 + 2], go = acc[mt * 4 + 3];
#pragma unroll
            for (int e = 0; e < 8; ++e) {
                const float cs = sigf(gf[e]) * creg[mt][e] + sigf(gi[e]) * tanhf(gg[e]);
                const float hv = sigf(go[e]) * tanhf(cs);
                creg[mt][e] = cs;
                const int row = mt * 16 + e + hi8;
                const int col = j0 + c16;
                hs[row * HS + col] = (bf16_t)hv;
                if (t == TT - 1) {
                    hfin[(size_t)(b0 + row) * HH + col] = (bf16_t)hv;
                    cfin[(size_t)(b0 + row) * HH + col] = cs;
                }
            }
        }
        // stage x_{t+1} (+ prefetch t+2 rss rows into L2)
        if (t + 1 < TT) {
            const int tn = t + 1;
            for (int idx = tid; idx < BTILE * XIN; idx += blockDim.x) {
                const int r = idx / XIN, cc = idx - r * XIN;
                const int b = b0 + r;
                const float* p = (cc < NI) ? &imu[((size_t)b * TT + tn) * NI + cc]
                                           : &rss[((size_t)b * TT + tn) * NA + (cc - NI)];
                xs[r * XS + cc] = (bf16_t)(*p);
                if (tn + 1 < TT) {
                    const float* pn = (cc < NI) ? &imu[((size_t)b * TT + tn + 1) * NI + cc]
                                                : &rss[((size_t)b * TT + tn + 1) * NA + (cc - NI)];
                    __builtin_prefetch(pn, 0, 1);
                }
            }
        }
        __syncthreads();
    }
}

// ---------------------------------------------------------------------------
// Kernel 3: decoder LSTMCell (10 steps, K = 256(h) + 2(dec_in) padded to 288)
// plus the H->2 output projection each step. Same wave mapping as encoder.
// ---------------------------------------------------------------------------
__global__ void __launch_bounds__(512)
decoder_kernel(const float* __restrict__ dec_in,
               const bf16_t* __restrict__ Wdec, const float* __restrict__ bdec,
               const float* __restrict__ W_out, const float* __restrict__ b_out,
               const float* __restrict__ cfin, const bf16_t* __restrict__ hfin,
               float* __restrict__ outp) {
    __shared__ __align__(16) bf16_t as_[BTILE * AS];  // [h(256) | x(2) | pad] bf16

    const int tid  = threadIdx.x;
    const int b0   = blockIdx.x * BTILE;
    const int lane = tid & 31;
    const int wv   = tid >> 5;
    const int j0   = wv * 16;
    const int c16  = lane & 15;
    const int hi8  = (lane >> 4) * 8;

    for (int i = tid; i < BTILE * AS; i += blockDim.x) as_[i] = (bf16_t)0.0f;
    __syncthreads();
    // stage h_f (encoder handoff) and x_0
    for (int idx = tid; idx < BTILE * HH; idx += blockDim.x) {
        const int r = idx >> 8, j = idx & (HH - 1);
        as_[r * AS + j] = hfin[(size_t)(b0 + r) * HH + j];
    }
    if (tid < BTILE * NCC) {
        const int r = tid >> 1, q = tid & 1;
        as_[r * AS + HH + q] = (bf16_t)dec_in[((size_t)(b0 + r) * OT) * NCC + q];
    }
    __syncthreads();

    float biasv[4];
#pragma unroll
    for (int g = 0; g < 4; ++g) biasv[g] = bdec[g * HH + j0 + c16];

    v8f creg[2];
#pragma unroll
    for (int mt = 0; mt < 2; ++mt)
#pragma unroll
        for (int e = 0; e < 8; ++e)
            creg[mt][e] = cfin[(size_t)(b0 + mt * 16 + e + hi8) * HH + j0 + c16];

    unsigned zoff = 0;

    for (int t = 0; t < OT; ++t) {
        asm volatile("" : "+s"(zoff));

        v8f acc[8];
#pragma unroll
        for (int mt = 0; mt < 2; ++mt)
#pragma unroll
            for (int g = 0; g < 4; ++g)
#pragma unroll
                for (int e = 0; e < 8; ++e) acc[mt * 4 + g][e] = biasv[g];

        for (int k0 = 0; k0 < DK; k0 += 32) {   // 9 WMMA k-steps x 8 WMMAs
            Frag fa[2];
#pragma unroll
            for (int mt = 0; mt < 2; ++mt) {
                const bf16_t* pa = &as_[(mt * 16 + c16) * AS + k0 + hi8];
                fa[mt].q[0] = *(const uint4*)pa;
                fa[mt].q[1] = *(const uint4*)(pa + 16);
            }
            Frag fb[4];
#pragma unroll
            for (int g = 0; g < 4; ++g) {
                const bf16_t* pb = Wdec + zoff +
                    (size_t)(g * HH + j0 + c16) * DK + k0 + hi8;
                fb[g].q[0] = *(const uint4*)pb;
                fb[g].q[1] = *(const uint4*)(pb + 16);
            }
#pragma unroll
            for (int g = 0; g < 4; ++g) {
#pragma unroll
                for (int mt = 0; mt < 2; ++mt)
                    acc[mt * 4 + g] = __builtin_amdgcn_wmma_f32_16x16x32_bf16(
                        false, fa[mt].v, false, fb[g].v, (short)0,
                        acc[mt * 4 + g], false, false);
            }
        }
        __syncthreads();

#pragma unroll
        for (int mt = 0; mt < 2; ++mt) {
            const v8f gi = acc[mt * 4 + 0], gf = acc[mt * 4 + 1];
            const v8f gg = acc[mt * 4 + 2], go = acc[mt * 4 + 3];
#pragma unroll
            for (int e = 0; e < 8; ++e) {
                const float cs = sigf(gf[e]) * creg[mt][e] + sigf(gi[e]) * tanhf(gg[e]);
                const float hv = sigf(go[e]) * tanhf(cs);
                creg[mt][e] = cs;
                as_[(mt * 16 + e + hi8) * AS + j0 + c16] = (bf16_t)hv;
            }
        }
        if (t + 1 < OT && tid < BTILE * NCC) {
            const int r = tid >> 1, q = tid & 1;
            as_[r * AS + HH + q] =
                (bf16_t)dec_in[((size_t)(b0 + r) * OT + t + 1) * NCC + q];
        }
        __syncthreads();

        // out_proj: pos[:, t] = h_{t+1} @ W_out^T + b_out   (64 threads, 2 cols)
        if (tid < BTILE * NCC) {
            const int r = tid >> 1, q = tid & 1;
            float s = b_out[q];
            for (int j = 0; j < HH; ++j)
                s += (float)as_[r * AS + j] * W_out[q * HH + j];
            outp[((size_t)(b0 + r) * OT + t) * NCC + q] = s;
        }
    }
}

// ---------------------------------------------------------------------------
extern "C" void kernel_launch(void* const* d_in, const int* in_sizes, int n_in,
                              void* d_out, int out_size, void* d_ws, size_t ws_size,
                              hipStream_t stream) {
    const float* imu    = (const float*)d_in[0];
    const float* rss    = (const float*)d_in[1];
    const float* dec_in = (const float*)d_in[2];
    const float* W_rss  = (const float*)d_in[3];
    const float* b_rss  = (const float*)d_in[4];
    const float* W_ih   = (const float*)d_in[5];
    const float* W_hh   = (const float*)d_in[6];
    const float* b_ih   = (const float*)d_in[7];
    const float* b_hh   = (const float*)d_in[8];
    const float* Wc_ih  = (const float*)d_in[9];
    const float* Wc_hh  = (const float*)d_in[10];
    const float* bc_ih  = (const float*)d_in[11];
    const float* bc_hh  = (const float*)d_in[12];
    const float* W_out  = (const float*)d_in[13];
    const float* b_out  = (const float*)d_in[14];
    float* outp = (float*)d_out;

    char* w = (char*)d_ws;
    bf16_t* WxP  = (bf16_t*)w; w += (size_t)G4 * XK * 2;   // 458752
    bf16_t* Whh  = (bf16_t*)w; w += (size_t)G4 * HH * 2;   // 524288
    bf16_t* Wdec = (bf16_t*)w; w += (size_t)G4 * DK * 2;   // 589824
    float*  beff = (float*)w;  w += (size_t)G4 * 4;        // 4096
    float*  bdec = (float*)w;  w += (size_t)G4 * 4;        // 4096
    bf16_t* hfin = (bf16_t*)w; w += (size_t)BB * HH * 2;   // 524288
    float*  cfin = (float*)w;  w += (size_t)BB * HH * 4;   // 1 MB
    if ((size_t)(w - (char*)d_ws) > ws_size) return;       // scratch too small

    prep_kernel<<<G4, 256, 0, stream>>>(W_rss, b_rss, W_ih, W_hh, b_ih, b_hh,
                                        Wc_ih, Wc_hh, bc_ih, bc_hh,
                                        WxP, Whh, Wdec, beff, bdec);
    encoder_kernel<<<NWG, 512, 0, stream>>>(imu, rss, WxP, Whh, beff, hfin, cfin);
    decoder_kernel<<<NWG, 512, 0, stream>>>(dec_in, Wdec, bdec, W_out, b_out,
                                            cfin, hfin, outp);
}
